// Seq2SeqAttentionModel_32804960206921
// MI455X (gfx1250) — compile-verified, matrix operand order checked by reading, not run on hardware
//
#include <hip/hip_runtime.h>
#include <hip/hip_bf16.h>
#include <cstdint>

// ---------------------------------------------------------------------------
// Seq2Seq LSTM + attention for MI455X (gfx1250, wave32, WMMA f16->f32).
// One workgroup = 16 batch rows, loops timesteps internally; recurrent state
// stays on-chip (h: f16 LDS for WMMA A-frags, c: f32 registers).
// GEMM inner loops batch-preload all 8 B-fragments per k-slice (one clause,
// one wait) then issue 8 independent back-to-back WMMAs.
// ---------------------------------------------------------------------------

#define B_    512
#define S_    336
#define IN_   7
#define H_    512
#define T_    96
#define G4_   2048   // 4*H
#define BT_   16     // batch tile per block
#define NWAVE 16
#define NTHR  512

typedef __attribute__((ext_vector_type(16))) _Float16 v16h;
typedef __attribute__((ext_vector_type(8)))  _Float16 v8h;
typedef __attribute__((ext_vector_type(8)))  float    v8f;

__device__ __forceinline__ float sigm(float x) { return 1.0f / (1.0f + __expf(-x)); }

// Load one 16-half WMMA fragment slice for this lane: 8 halves at p, 8 at p+16.
// p already includes row base + k0 + (lane>>4)*8.
__device__ __forceinline__ v16h frag_ld(const _Float16* p) {
  v8h lo = *(const v8h*)(p);
  v8h hi = *(const v8h*)(p + 16);
  return __builtin_shufflevector(lo, hi, 0,1,2,3,4,5,6,7,8,9,10,11,12,13,14,15);
}

__device__ __forceinline__ v8f wmma_f16(v16h a, v16h b, v8f c) {
  return __builtin_amdgcn_wmma_f32_16x16x32_f16(false, a, false, b, (short)0, c,
                                                false, false);
}

// ---------------------------------------------------------------------------
// f32 -> f16 weight conversion
// ---------------------------------------------------------------------------
__global__ void cvt_f32_f16(const float* __restrict__ src,
                            _Float16* __restrict__ dst, int n) {
  int i = blockIdx.x * 256 + threadIdx.x;
  if (i < n) dst[i] = (_Float16)src[i];
}

// ---------------------------------------------------------------------------
// Fused 2-layer LSTM encoder. Grid: 32 blocks x 512 threads.
// ---------------------------------------------------------------------------
__global__ __launch_bounds__(NTHR)
void enc_kernel(const float* __restrict__ x,          // (B,S,IN) f32
                const float* __restrict__ w_ih0,      // (2048,7) f32
                const float* __restrict__ b0,         // (2048) f32
                const float* __restrict__ b1,         // (2048) f32
                const _Float16* __restrict__ whh0,    // (2048,512) f16
                const _Float16* __restrict__ wih1,    // (2048,512) f16
                const _Float16* __restrict__ whh1,    // (2048,512) f16
                _Float16* __restrict__ enc_out,       // (B,S,512) f16
                float* __restrict__ hfin,             // (B,512) f32
                float* __restrict__ cfin)             // (B,512) f32
{
  __shared__ _Float16 h0s[BT_ * H_];   // 16 KB
  __shared__ _Float16 h1s[BT_ * H_];   // 16 KB
  __shared__ float    xs[BT_ * 8];     // staged x_t

  const int tid   = threadIdx.x;
  const int wave  = tid >> 5;
  const int lane  = tid & 31;
  const int vcol  = lane & 15;        // N-within-tile / A-row
  const int mlo   = (lane >> 4) * 8;  // K-half select * 8 / C-frag row offset
  const int b0idx = blockIdx.x * BT_;

  float c0r[2][8], c1r[2][8];
  #pragma unroll
  for (int g = 0; g < 2; ++g)
    #pragma unroll
    for (int r = 0; r < 8; ++r) { c0r[g][r] = 0.f; c1r[g][r] = 0.f; }

  for (int i = tid; i < BT_ * H_; i += NTHR) {
    h0s[i] = (_Float16)0.f; h1s[i] = (_Float16)0.f;
  }
  __syncthreads();

  for (int t = 0; t < S_; ++t) {
    // stage x_t (16 rows x 7)
    if (tid < BT_ * IN_) {
      int m = tid / IN_, k = tid % IN_;
      xs[m * 8 + k] = x[(size_t)(b0idx + m) * S_ * IN_ + (size_t)t * IN_ + k];
    }
    __syncthreads();

    // ---------------- layer 0 gates: b0 + x@Wih0^T (VALU) + h0@Whh0^T (WMMA)
    v8f acc[8];
    #pragma unroll
    for (int j = 0; j < 8; ++j) {
      const int col = 16 * wave + 256 * j + vcol;
      const float bias = b0[col];
      const float* wr = &w_ih0[col * IN_];
      float w0 = wr[0], w1 = wr[1], w2 = wr[2], w3 = wr[3],
            w4 = wr[4], w5 = wr[5], w6 = wr[6];
      #pragma unroll
      for (int r = 0; r < 8; ++r) {
        const int m = r + mlo;
        float s = bias;
        s += xs[m*8+0]*w0 + xs[m*8+1]*w1 + xs[m*8+2]*w2 + xs[m*8+3]*w3;
        s += xs[m*8+4]*w4 + xs[m*8+5]*w5 + xs[m*8+6]*w6;
        acc[j][r] = s;
      }
    }
    for (int k0 = 0; k0 < H_; k0 += 32) {
      v16h af = frag_ld(&h0s[vcol * H_ + k0 + mlo]);
      v16h bf[8];
      #pragma unroll
      for (int j = 0; j < 8; ++j)
        bf[j] = frag_ld(&whh0[(size_t)(16 * wave + 256 * j + vcol) * H_ + k0 + mlo]);
      #pragma unroll
      for (int j = 0; j < 8; ++j)
        acc[j] = wmma_f16(af, bf[j], acc[j]);
    }
    __syncthreads();   // all reads of h0s done

    // layer-0 cell update (wave-local gates; c in registers, h -> LDS f16)
    #pragma unroll
    for (int g = 0; g < 2; ++g) {
      const int u = 16 * wave + 256 * g + vcol;
      #pragma unroll
      for (int r = 0; r < 8; ++r) {
        const int m = r + mlo;
        const float cn = sigm(acc[2+g][r]) * c0r[g][r]
                       + sigm(acc[0+g][r]) * tanhf(acc[4+g][r]);
        const float hn = sigm(acc[6+g][r]) * tanhf(cn);
        c0r[g][r] = cn;
        h0s[m * H_ + u] = (_Float16)hn;
      }
    }
    __syncthreads();   // h0_new visible

    // ---------------- layer 1 gates: b1 + h0@Wih1^T + h1@Whh1^T (both WMMA)
    #pragma unroll
    for (int j = 0; j < 8; ++j) {
      const float bias = b1[16 * wave + 256 * j + vcol];
      #pragma unroll
      for (int r = 0; r < 8; ++r) acc[j][r] = bias;
    }
    for (int k0 = 0; k0 < H_; k0 += 32) {   // ih1 chain
      v16h a0 = frag_ld(&h0s[vcol * H_ + k0 + mlo]);
      v16h bf[8];
      #pragma unroll
      for (int j = 0; j < 8; ++j)
        bf[j] = frag_ld(&wih1[(size_t)(16 * wave + 256 * j + vcol) * H_ + k0 + mlo]);
      #pragma unroll
      for (int j = 0; j < 8; ++j)
        acc[j] = wmma_f16(a0, bf[j], acc[j]);
    }
    for (int k0 = 0; k0 < H_; k0 += 32) {   // hh1 chain
      v16h a1 = frag_ld(&h1s[vcol * H_ + k0 + mlo]);
      v16h bf[8];
      #pragma unroll
      for (int j = 0; j < 8; ++j)
        bf[j] = frag_ld(&whh1[(size_t)(16 * wave + 256 * j + vcol) * H_ + k0 + mlo]);
      #pragma unroll
      for (int j = 0; j < 8; ++j)
        acc[j] = wmma_f16(a1, bf[j], acc[j]);
    }
    __syncthreads();   // all reads of h1s done

    // layer-1 cell update
    #pragma unroll
    for (int g = 0; g < 2; ++g) {
      const int u = 16 * wave + 256 * g + vcol;
      #pragma unroll
      for (int r = 0; r < 8; ++r) {
        const int m = r + mlo;
        const float cn = sigm(acc[2+g][r]) * c1r[g][r]
                       + sigm(acc[0+g][r]) * tanhf(acc[4+g][r]);
        const float hn = sigm(acc[6+g][r]) * tanhf(cn);
        c1r[g][r] = cn;
        h1s[m * H_ + u] = (_Float16)hn;
        if (t == S_ - 1) {
          hfin[(size_t)(b0idx + m) * H_ + u] = hn;
          cfin[(size_t)(b0idx + m) * H_ + u] = cn;
        }
      }
    }
    __syncthreads();   // h1_new visible

    // stream enc_out_t = h1 (f16, coalesced)
    for (int i = tid; i < BT_ * H_; i += NTHR) {
      int m = i >> 9, u = i & (H_ - 1);
      enc_out[((size_t)(b0idx + m) * S_ + t) * H_ + u] = h1s[i];
    }
    __syncthreads();
  }
}

// ---------------------------------------------------------------------------
// Attention decoder: 96 steps. Grid: 32 blocks x 512 threads.
// ---------------------------------------------------------------------------
__global__ __launch_bounds__(NTHR)
void dec_kernel(const float* __restrict__ x,          // (B,S,IN) f32 (for di0)
                const float* __restrict__ dec_W_ih,   // (2048,1) f32
                const float* __restrict__ dec_b,      // (2048) f32
                const float* __restrict__ cat_b,      // (512) f32
                const float* __restrict__ fc_W,       // (1,512) f32
                const float* __restrict__ fc_b,       // (1) f32
                const _Float16* __restrict__ dwhh,    // (2048,512) f16
                const _Float16* __restrict__ attw,    // (512,512) f16
                const _Float16* __restrict__ catw,    // (512,1024) f16
                const _Float16* __restrict__ enc_out, // (B,S,512) f16
                const float* __restrict__ hfin,
                const float* __restrict__ cfin,
                float* __restrict__ out)              // (B,T) f32
{
  __shared__ _Float16 hc[BT_ * 1024];  // [h | ctx] f16, 32 KB
  __shared__ _Float16 sc[BT_ * S_];    // scores -> softmax weights, 10.5 KB
  __shared__ _Float16 tmp[BT_ * H_];   // q then cc, 16 KB
  __shared__ float    dis[BT_];

  const int tid   = threadIdx.x;
  const int wave  = tid >> 5;
  const int lane  = tid & 31;
  const int vcol  = lane & 15;
  const int mlo   = (lane >> 4) * 8;
  const int b0idx = blockIdx.x * BT_;

  float cr[2][8];
  #pragma unroll
  for (int g = 0; g < 2; ++g) {
    const int u = 16 * wave + 256 * g + vcol;
    #pragma unroll
    for (int r = 0; r < 8; ++r)
      cr[g][r] = cfin[(size_t)(b0idx + r + mlo) * H_ + u];
  }
  for (int i = tid; i < BT_ * 1024; i += NTHR) {
    int m = i >> 10, u = i & 1023;
    hc[i] = (u < H_) ? (_Float16)hfin[(size_t)(b0idx + m) * H_ + u]
                     : (_Float16)0.f;
  }
  if (tid < BT_)
    dis[tid] = x[(size_t)(b0idx + tid) * S_ * IN_ + (size_t)(S_ - 1) * IN_ + (IN_ - 1)];
  __syncthreads();

  for (int t = 0; t < T_; ++t) {
    // ---------------- LSTM cell gates: dec_b + di*Wih + h@Whh^T (WMMA)
    v8f acc[8];
    #pragma unroll
    for (int j = 0; j < 8; ++j) {
      const int col = 16 * wave + 256 * j + vcol;
      const float bias = dec_b[col];
      const float wih  = dec_W_ih[col];
      #pragma unroll
      for (int r = 0; r < 8; ++r) acc[j][r] = bias + dis[r + mlo] * wih;
    }
    for (int k0 = 0; k0 < H_; k0 += 32) {
      v16h af = frag_ld(&hc[vcol * 1024 + k0 + mlo]);
      v16h bf[8];
      #pragma unroll
      for (int j = 0; j < 8; ++j)
        bf[j] = frag_ld(&dwhh[(size_t)(16 * wave + 256 * j + vcol) * H_ + k0 + mlo]);
      #pragma unroll
      for (int j = 0; j < 8; ++j)
        acc[j] = wmma_f16(af, bf[j], acc[j]);
    }
    __syncthreads();

    #pragma unroll
    for (int g = 0; g < 2; ++g) {
      const int u = 16 * wave + 256 * g + vcol;
      #pragma unroll
      for (int r = 0; r < 8; ++r) {
        const int m = r + mlo;
        const float cn = sigm(acc[2+g][r]) * cr[g][r]
                       + sigm(acc[0+g][r]) * tanhf(acc[4+g][r]);
        const float hn = sigm(acc[6+g][r]) * tanhf(cn);
        cr[g][r] = cn;
        hc[m * 1024 + u] = (_Float16)hn;
      }
    }
    __syncthreads();

    // ---------------- q = h @ att_W^T  (WMMA, 2 N-tiles per wave) -> tmp f16
    {
      v8f qa0 = {}, qa1 = {};
      for (int k0 = 0; k0 < H_; k0 += 32) {
        v16h af = frag_ld(&hc[vcol * 1024 + k0 + mlo]);
        v16h bf0 = frag_ld(&attw[(size_t)(16*wave       + vcol) * H_ + k0 + mlo]);
        v16h bf1 = frag_ld(&attw[(size_t)(16*wave + 256 + vcol) * H_ + k0 + mlo]);
        qa0 = wmma_f16(af, bf0, qa0);
        qa1 = wmma_f16(af, bf1, qa1);
      }
      #pragma unroll
      for (int r = 0; r < 8; ++r) {
        tmp[(r + mlo) * H_ + 16*wave       + vcol] = (_Float16)qa0[r];
        tmp[(r + mlo) * H_ + 16*wave + 256 + vcol] = (_Float16)qa1[r];
      }
    }
    __syncthreads();

    // ---------------- scores[b][s] = enc[b][s] . q[b]  (VALU)
    for (int p = tid; p < BT_ * S_; p += NTHR) {
      const int m = p / S_, s = p % S_;
      const _Float16* er = &enc_out[((size_t)(b0idx + m) * S_ + s) * H_];
      const _Float16* qr = &tmp[m * H_];
      float a = 0.f;
      for (int k = 0; k < H_; k += 8) {
        v8h e = *(const v8h*)(er + k);
        v8h q = *(const v8h*)(qr + k);
        #pragma unroll
        for (int ii = 0; ii < 8; ++ii) a += (float)e[ii] * (float)q[ii];
      }
      sc[m * S_ + s] = (_Float16)a;
    }
    __syncthreads();

    // ---------------- softmax over s (wave w -> batch row w)
    {
      const int m = wave;
      float mx = -1e30f;
      for (int s = lane; s < S_; s += 32) mx = fmaxf(mx, (float)sc[m * S_ + s]);
      #pragma unroll
      for (int d = 16; d > 0; d >>= 1) mx = fmaxf(mx, __shfl_xor(mx, d));
      float sum = 0.f;
      for (int s = lane; s < S_; s += 32) sum += __expf((float)sc[m * S_ + s] - mx);
      #pragma unroll
      for (int d = 16; d > 0; d >>= 1) sum += __shfl_xor(sum, d);
      const float inv = 1.0f / sum;
      for (int s = lane; s < S_; s += 32)
        sc[m * S_ + s] = (_Float16)(__expf((float)sc[m * S_ + s] - mx) * inv);
    }
    __syncthreads();

    // ---------------- ctx[b] = sum_s w[b][s] * enc[b][s]  -> hc[:,512:]
    {
      const int m = tid >> 5;
      float av[16];
      #pragma unroll
      for (int i = 0; i < 16; ++i) av[i] = 0.f;
      for (int s = 0; s < S_; ++s) {
        const float wgt = (float)sc[m * S_ + s];
        const _Float16* er = &enc_out[((size_t)(b0idx + m) * S_ + s) * H_];
        #pragma unroll
        for (int i = 0; i < 16; ++i) av[i] += wgt * (float)er[lane + 32 * i];
      }
      #pragma unroll
      for (int i = 0; i < 16; ++i)
        hc[m * 1024 + H_ + lane + 32 * i] = (_Float16)av[i];
    }
    __syncthreads();

    // ---------------- cc = tanh([h|ctx] @ cat_W^T + cat_b)  (WMMA, K=1024)
    {
      v8f ca0 = {}, ca1 = {};
      for (int k0 = 0; k0 < 1024; k0 += 32) {
        v16h af = frag_ld(&hc[vcol * 1024 + k0 + mlo]);
        v16h bf0 = frag_ld(&catw[(size_t)(16*wave       + vcol) * 1024 + k0 + mlo]);
        v16h bf1 = frag_ld(&catw[(size_t)(16*wave + 256 + vcol) * 1024 + k0 + mlo]);
        ca0 = wmma_f16(af, bf0, ca0);
        ca1 = wmma_f16(af, bf1, ca1);
      }
      const int u0 = 16 * wave + vcol, u1 = u0 + 256;
      #pragma unroll
      for (int r = 0; r < 8; ++r) {
        tmp[(r + mlo) * H_ + u0] = (_Float16)tanhf(ca0[r] + cat_b[u0]);
        tmp[(r + mlo) * H_ + u1] = (_Float16)tanhf(ca1[r] + cat_b[u1]);
      }
    }
    __syncthreads();

    // ---------------- pred = cc @ fc_W^T + fc_b  (wave w -> batch row w)
    {
      const int m = wave;
      float s = 0.f;
      for (int k = lane; k < H_; k += 32) s += (float)tmp[m * H_ + k] * fc_W[k];
      #pragma unroll
      for (int d = 16; d > 0; d >>= 1) s += __shfl_xor(s, d);
      if (lane == 0) {
        const float p = s + fc_b[0];
        out[(size_t)(b0idx + m) * T_ + t] = p;
        dis[m] = p;
      }
    }
    __syncthreads();
  }
}

// ---------------------------------------------------------------------------
// Launch
// ---------------------------------------------------------------------------
extern "C" void kernel_launch(void* const* d_in, const int* in_sizes, int n_in,
                              void* d_out, int out_size, void* d_ws, size_t ws_size,
                              hipStream_t stream) {
  const float* x     = (const float*)d_in[0];
  const float* wih0  = (const float*)d_in[1];
  const float* whh0f = (const float*)d_in[2];
  const float* b0    = (const float*)d_in[3];
  const float* wih1f = (const float*)d_in[4];
  const float* whh1f = (const float*)d_in[5];
  const float* b1    = (const float*)d_in[6];
  const float* dwihf = (const float*)d_in[7];
  const float* dwhhf = (const float*)d_in[8];
  const float* db    = (const float*)d_in[9];
  const float* attwf = (const float*)d_in[10];
  const float* catwf = (const float*)d_in[11];
  const float* catb  = (const float*)d_in[12];
  const float* fcw   = (const float*)d_in[13];
  const float* fcb   = (const float*)d_in[14];
  float* out = (float*)d_out;

  // workspace layout (bytes)
  constexpr size_t ENC_B = (size_t)B_ * S_ * H_ * 2;   // 176,160,768
  constexpr size_t W_B   = (size_t)G4_ * H_ * 2;       // 2,097,152
  constexpr size_t ATT_B = (size_t)H_ * H_ * 2;
  constexpr size_t CAT_B = (size_t)H_ * 2 * H_ * 2;
  constexpr size_t FIN_B = (size_t)B_ * H_ * 4;

  char* ws = (char*)d_ws;
  _Float16* enc  = (_Float16*)(ws);
  _Float16* whh0 = (_Float16*)(ws + ENC_B);
  _Float16* wih1 = (_Float16*)(ws + ENC_B + 1 * W_B);
  _Float16* whh1 = (_Float16*)(ws + ENC_B + 2 * W_B);
  _Float16* dwhh = (_Float16*)(ws + ENC_B + 3 * W_B);
  _Float16* attw = (_Float16*)(ws + ENC_B + 4 * W_B);
  _Float16* catw = (_Float16*)(ws + ENC_B + 4 * W_B + ATT_B);
  float*    hfin = (float*)   (ws + ENC_B + 4 * W_B + ATT_B + CAT_B);
  float*    cfin = (float*)   (ws + ENC_B + 4 * W_B + ATT_B + CAT_B + FIN_B);

  const int nw = G4_ * H_;
  cvt_f32_f16<<<(nw + 255) / 256, 256, 0, stream>>>(whh0f, whh0, nw);
  cvt_f32_f16<<<(nw + 255) / 256, 256, 0, stream>>>(wih1f, wih1, nw);
  cvt_f32_f16<<<(nw + 255) / 256, 256, 0, stream>>>(whh1f, whh1, nw);
  cvt_f32_f16<<<(nw + 255) / 256, 256, 0, stream>>>(dwhhf, dwhh, nw);
  const int na = H_ * H_;
  cvt_f32_f16<<<(na + 255) / 256, 256, 0, stream>>>(attwf, attw, na);
  const int nc = H_ * 2 * H_;
  cvt_f32_f16<<<(nc + 255) / 256, 256, 0, stream>>>(catwf, catw, nc);

  enc_kernel<<<B_ / BT_, NTHR, 0, stream>>>(x, wih0, b0, b1, whh0, wih1, whh1,
                                            enc, hfin, cfin);
  dec_kernel<<<B_ / BT_, NTHR, 0, stream>>>(x, dwihf, db, catb, fcw, fcb,
                                            dwhh, attw, catw, enc, hfin, cfin,
                                            out);
}
